// MultiHeadAttention_46239617908907
// MI455X (gfx1250) — matrix operand order repaired and checked by reading, not compiled, not run on hardware
//
#include <hip/hip_runtime.h>
#include <hip/hip_bf16.h>

// ---------------------------------------------------------------------------
// MHA for MI455X (gfx1250): f16 WMMA pipeline, flash-attention inner kernel.
// Shapes: B=2, S=2048, D=1024, H=16, Dh=64.
// ---------------------------------------------------------------------------

typedef __attribute__((ext_vector_type(16))) _Float16 v16h;
typedef __attribute__((ext_vector_type(8)))  _Float16 v8h;
typedef __attribute__((ext_vector_type(8)))  float    v8f;

union Frag16 { v16h v; v8h h[2]; };

#define D_MODEL 1024
#define SEQ     2048
#define NHEAD   16
#define DHEAD   64
#define BATCH   2
#define MTOT    (BATCH * SEQ)      // 4096 rows

__device__ __forceinline__ v8f wmma_f16(v16h a, v16h b, v8f c) {
  // D = A(16x32 f16) * B(32x16 f16) + C(16x16 f32)
  return __builtin_amdgcn_wmma_f32_16x16x32_f16(
      /*neg_a=*/false, a, /*neg_b=*/false, b,
      /*c_mod=*/(short)0, c, /*reuse_a=*/false, /*reuse_b=*/false);
}

// ---------------------------------------------------------------------------
// fp32 -> f16 conversion (grid-stride elementwise)
// ---------------------------------------------------------------------------
__global__ void cvt_f32_to_f16(const float* __restrict__ in,
                               _Float16* __restrict__ out, int n) {
  int i = blockIdx.x * blockDim.x + threadIdx.x;
  int stride = gridDim.x * blockDim.x;
  for (; i < n; i += stride) out[i] = (_Float16)in[i];
}

// ---------------------------------------------------------------------------
// Epilogue helper: store one 16x16 f32 C-fragment per mode.
//   mode 0: f16 flat [M, D_MODEL] with (acc+bias)*scale   (Q scaled, K)
//   mode 1: f16 transposed [B,H,Dh,S]                     (V)
//   mode 2: f32 flat [M, D_MODEL]                         (output projection)
// ---------------------------------------------------------------------------
__device__ __forceinline__ void store_tile(v8f acc, int m0, int ncol,
                                           int rbase, float bv, float scale,
                                           void* out, int mode) {
  if (mode == 0) {
    _Float16* o = (_Float16*)out;
#pragma unroll
    for (int r = 0; r < 8; ++r) {
      float v = (acc[r] + bv) * scale;
      o[(size_t)(m0 + rbase + r) * D_MODEL + ncol] = (_Float16)v;
    }
  } else if (mode == 1) {
    _Float16* o = (_Float16*)out;                // Vt[B][H][Dh][S]
    int hh = ncol >> 6, dd = ncol & (DHEAD - 1);
#pragma unroll
    for (int r = 0; r < 8; ++r) {
      int m = m0 + rbase + r;
      int bb = m >> 11, st = m & (SEQ - 1);      // m = bb*SEQ + st
      float v = (acc[r] + bv) * scale;
      o[(((size_t)(bb * NHEAD + hh) * DHEAD + dd) << 11) + st] = (_Float16)v;
    }
  } else {
    float* o = (float*)out;
#pragma unroll
    for (int r = 0; r < 8; ++r) {
      o[(size_t)(m0 + rbase + r) * D_MODEL + ncol] = (acc[r] + bv) * scale;
    }
  }
}

// ---------------------------------------------------------------------------
// C[M,N] = (A[M,K] @ W[N,K]^T + bias) * scale
// One wave per 16x64 output tile (4 accumulators). Fragments for k-step
// k0+32 are loaded into a second register set BEFORE the 4 WMMAs of k0
// (double-buffered software pipeline), so matrix ops overlap VMEM instead
// of draining LOADcnt to 0 before every WMMA.
// ---------------------------------------------------------------------------
__global__ void __launch_bounds__(256) proj_gemm(
    const _Float16* __restrict__ A, const _Float16* __restrict__ W,
    const float* __restrict__ bias, void* __restrict__ out,
    float scale, int mode) {
  const int NGRP = D_MODEL / 64;                               // 16 n-groups
  int wave = (int)((blockIdx.x * blockDim.x + threadIdx.x) >> 5);
  int lane = threadIdx.x & 31;
  int mt = wave / NGRP, ng = wave - mt * NGRP;
  int m0 = mt << 4, n0 = ng << 6;                              // 16 x 64 tile
  int lrow = lane & 15;
  int hi = (lane >> 4) & 1;

  // A fragment: lane L<16 -> row m0+L, K = k0..k0+15 contig
  //             lane L>=16 -> row m0+L-16, K = k0+16..k0+31 contig
  const _Float16* Ap = A + (size_t)(m0 + lrow) * D_MODEL + (hi ? 16 : 0);
  // B fragment from W[N,K] row-major: lane n<16 -> col n0+n, K=k0..k0+15 contig
  const _Float16* Wp = W + (size_t)(n0 + lrow) * D_MODEL + (hi ? 16 : 0);

  v8f acc0 = {}, acc1 = {}, acc2 = {}, acc3 = {};

  Frag16 af[2], bf[2][4];
  // prologue: stage k0 = 0 into buffer 0
  af[0].h[0] = *(const v8h*)(Ap);
  af[0].h[1] = *(const v8h*)(Ap + 8);
  bf[0][0].v = *(const v16h*)(Wp);
  bf[0][1].v = *(const v16h*)(Wp + 16 * D_MODEL);
  bf[0][2].v = *(const v16h*)(Wp + 32 * D_MODEL);
  bf[0][3].v = *(const v16h*)(Wp + 48 * D_MODEL);

#pragma unroll
  for (int k0 = 0; k0 < D_MODEL; k0 += 32) {
    const int cur = (k0 >> 5) & 1;
    const int nxt = cur ^ 1;
    const int k1 = k0 + 32;
    if (k1 < D_MODEL) {                           // stage next k-step
      __builtin_prefetch(Ap + k1 + 256, 0, 1);    // global_prefetch_b8
      __builtin_prefetch(Wp + k1 + 256, 0, 1);
      af[nxt].h[0] = *(const v8h*)(Ap + k1);
      af[nxt].h[1] = *(const v8h*)(Ap + k1 + 8);
      bf[nxt][0].v = *(const v16h*)(Wp + k1);
      bf[nxt][1].v = *(const v16h*)(Wp + 16 * D_MODEL + k1);
      bf[nxt][2].v = *(const v16h*)(Wp + 32 * D_MODEL + k1);
      bf[nxt][3].v = *(const v16h*)(Wp + 48 * D_MODEL + k1);
    }
    acc0 = wmma_f16(af[cur].v, bf[cur][0].v, acc0);
    acc1 = wmma_f16(af[cur].v, bf[cur][1].v, acc1);
    acc2 = wmma_f16(af[cur].v, bf[cur][2].v, acc2);
    acc3 = wmma_f16(af[cur].v, bf[cur][3].v, acc3);
  }

  int col = lrow;
  int rbase = hi ? 8 : 0;
  store_tile(acc0, m0, n0 + 0  + col, rbase, bias[n0 + 0  + col], scale, out, mode);
  store_tile(acc1, m0, n0 + 16 + col, rbase, bias[n0 + 16 + col], scale, out, mode);
  store_tile(acc2, m0, n0 + 32 + col, rbase, bias[n0 + 32 + col], scale, out, mode);
  store_tile(acc3, m0, n0 + 48 + col, rbase, bias[n0 + 48 + col], scale, out, mode);
}

// ---------------------------------------------------------------------------
// Flash attention: one wave (32 threads) per (b, h, 16-row query tile).
// Streams K/V in 32-column steps; online softmax; P restriped via LDS.
// Q was pre-scaled by Dh^-0.5 in its projection.
// ---------------------------------------------------------------------------
__global__ void __launch_bounds__(32) flash_attn(
    const _Float16* __restrict__ Qh, const _Float16* __restrict__ Kh,
    const _Float16* __restrict__ Vt, const float* __restrict__ mask,
    _Float16* __restrict__ AO) {
  __shared__ _Float16 ldsP[16 * 32];             // 1 KB probability tile

  int lane = threadIdx.x;
  int bid = blockIdx.x;
  int tt = bid & (SEQ / 16 - 1);                 // query tile 0..127
  int h  = (bid >> 7) & (NHEAD - 1);
  int b  = bid >> 11;
  int t0 = tt << 4;
  int lrow = lane & 15;
  int hi = (lane >> 4) & 1;
  int col = lrow;
  int rbase = hi ? 8 : 0;

  // Q fragments (16x64 = two A-frags of K=32), loaded once
  Frag16 qa0, qa1;
  {
    const _Float16* qp = Qh + (size_t)(b * SEQ + t0 + lrow) * D_MODEL
                            + h * DHEAD + (hi ? 16 : 0);
    qa0.h[0] = *(const v8h*)(qp);       qa0.h[1] = *(const v8h*)(qp + 8);
    qa1.h[0] = *(const v8h*)(qp + 32);  qa1.h[1] = *(const v8h*)(qp + 40);
  }
  const _Float16* Kbase = Kh + (size_t)(b * SEQ) * D_MODEL + h * DHEAD
                             + (hi ? 16 : 0);
  const _Float16* Vbase = Vt + ((size_t)(b * NHEAD + h) * DHEAD) * SEQ;
  const float*    Mbase = mask + (size_t)(b * SEQ + t0) * SEQ;

  float mrow[8], lsum[8];
  v8f o0 = {}, o1 = {}, o2 = {}, o3 = {};
#pragma unroll
  for (int r = 0; r < 8; ++r) { mrow[r] = -INFINITY; lsum[r] = 0.0f; }

  for (int s0 = 0; s0 < SEQ; s0 += 32) {
    // ---- scores: S = Q * K^T for two 16-col subtiles, K-dim = Dh = 64 ----
    v8f sc0 = {}, sc1 = {};
    {
      const _Float16* kp0 = Kbase + (size_t)(s0 + lrow) * D_MODEL;
      const _Float16* kp1 = Kbase + (size_t)(s0 + 16 + lrow) * D_MODEL;
      __builtin_prefetch(kp0 + 32 * D_MODEL, 0, 1);
      Frag16 kb;
      kb.v = *(const v16h*)(kp0);        sc0 = wmma_f16(qa0.v, kb.v, sc0);
      kb.v = *(const v16h*)(kp0 + 32);   sc0 = wmma_f16(qa1.v, kb.v, sc0);
      kb.v = *(const v16h*)(kp1);        sc1 = wmma_f16(qa0.v, kb.v, sc1);
      kb.v = *(const v16h*)(kp1 + 32);   sc1 = wmma_f16(qa1.v, kb.v, sc1);
    }
    // ---- additive attention mask ----
#pragma unroll
    for (int r = 0; r < 8; ++r) {
      const float* mp = Mbase + (size_t)(rbase + r) * SEQ + s0 + col;
      sc0[r] += mp[0];
      sc1[r] += mp[16];
    }
    // ---- online softmax update (row reductions within 16-lane groups) ----
#pragma unroll
    for (int r = 0; r < 8; ++r) {
      float tv = fmaxf(sc0[r], sc1[r]);
      tv = fmaxf(tv, __shfl_xor(tv, 1, 32));
      tv = fmaxf(tv, __shfl_xor(tv, 2, 32));
      tv = fmaxf(tv, __shfl_xor(tv, 4, 32));
      tv = fmaxf(tv, __shfl_xor(tv, 8, 32));
      float mn = fmaxf(mrow[r], tv);
      float al = __expf(mrow[r] - mn);
      mrow[r] = mn;
      float p0 = __expf(sc0[r] - mn);
      float p1 = __expf(sc1[r] - mn);
      float rs = p0 + p1;
      rs += __shfl_xor(rs, 1, 32);
      rs += __shfl_xor(rs, 2, 32);
      rs += __shfl_xor(rs, 4, 32);
      rs += __shfl_xor(rs, 8, 32);
      lsum[r] = lsum[r] * al + rs;
      o0[r] *= al; o1[r] *= al; o2[r] *= al; o3[r] *= al;
      // restripe P from C-layout to row-major 16x32 in LDS
      ldsP[(rbase + r) * 32 + col]      = (_Float16)p0;
      ldsP[(rbase + r) * 32 + 16 + col] = (_Float16)p1;
    }
    asm volatile("s_wait_dscnt 0" ::: "memory");   // LDS RAW fence (wave-local)

    // P as A-fragment (16x32 f16)
    Frag16 pa;
    {
      const _Float16* pp = &ldsP[lrow * 32 + (hi ? 16 : 0)];
      pa.h[0] = *(const v8h*)(pp);
      pa.h[1] = *(const v8h*)(pp + 8);
    }
    // ---- O += P * V  (Vt rows are contiguous over s => contig B-frags) ----
    {
      const _Float16* vp = Vbase + (size_t)lrow * SEQ + s0 + (hi ? 16 : 0);
      __builtin_prefetch(vp + 32, 0, 1);
      Frag16 vb;
      vb.v = *(const v16h*)(vp);             o0 = wmma_f16(pa.v, vb.v, o0);
      vb.v = *(const v16h*)(vp + 16 * SEQ);  o1 = wmma_f16(pa.v, vb.v, o1);
      vb.v = *(const v16h*)(vp + 32 * SEQ);  o2 = wmma_f16(pa.v, vb.v, o2);
      vb.v = *(const v16h*)(vp + 48 * SEQ);  o3 = wmma_f16(pa.v, vb.v, o3);
    }
  }

  // normalize and store f16 tile into AO[b*S+t, h*64+d]
  _Float16* op = AO + (size_t)(b * SEQ + t0) * D_MODEL + h * DHEAD;
#pragma unroll
  for (int r = 0; r < 8; ++r) {
    float inv = 1.0f / lsum[r];
    size_t rowoff = (size_t)(rbase + r) * D_MODEL;
    op[rowoff + 0  + col] = (_Float16)(o0[r] * inv);
    op[rowoff + 16 + col] = (_Float16)(o1[r] * inv);
    op[rowoff + 32 + col] = (_Float16)(o2[r] * inv);
    op[rowoff + 48 + col] = (_Float16)(o3[r] * inv);
  }
}

// ---------------------------------------------------------------------------
// Host-side launch
// ---------------------------------------------------------------------------
extern "C" void kernel_launch(void* const* d_in, const int* in_sizes, int n_in,
                              void* d_out, int out_size, void* d_ws, size_t ws_size,
                              hipStream_t stream) {
  (void)in_sizes; (void)n_in; (void)out_size; (void)ws_size;

  const float* q   = (const float*)d_in[0];
  const float* kv  = (const float*)d_in[1];
  const float* msk = (const float*)d_in[2];
  const float* Wq  = (const float*)d_in[3];
  const float* bq  = (const float*)d_in[4];
  const float* Wk  = (const float*)d_in[5];
  const float* bk  = (const float*)d_in[6];
  const float* Wv  = (const float*)d_in[7];
  const float* bv  = (const float*)d_in[8];
  const float* Wo  = (const float*)d_in[9];
  const float* bo  = (const float*)d_in[10];

  const size_t MD = (size_t)MTOT * D_MODEL;      // 4,194,304 elements
  const size_t DD = (size_t)D_MODEL * D_MODEL;   // 1,048,576 elements

  // workspace carve-out (~56 MB total)
  char* ws = (char*)d_ws;
  _Float16* Xq  = (_Float16*)ws; ws += MD * 2;
  _Float16* Xkv = (_Float16*)ws; ws += MD * 2;
  _Float16* Qh  = (_Float16*)ws; ws += MD * 2;
  _Float16* Kh  = (_Float16*)ws; ws += MD * 2;
  _Float16* Vt  = (_Float16*)ws; ws += MD * 2;   // transposed [B,H,Dh,S]
  _Float16* AO  = (_Float16*)ws; ws += MD * 2;
  _Float16* Wqh = (_Float16*)ws; ws += DD * 2;
  _Float16* Wkh = (_Float16*)ws; ws += DD * 2;
  _Float16* Wvh = (_Float16*)ws; ws += DD * 2;
  _Float16* Woh = (_Float16*)ws; ws += DD * 2;

  // fp32 -> f16 conversions
  cvt_f32_to_f16<<<1024, 256, 0, stream>>>(q,  Xq,  (int)MD);
  cvt_f32_to_f16<<<1024, 256, 0, stream>>>(kv, Xkv, (int)MD);
  cvt_f32_to_f16<<<512,  256, 0, stream>>>(Wq, Wqh, (int)DD);
  cvt_f32_to_f16<<<512,  256, 0, stream>>>(Wk, Wkh, (int)DD);
  cvt_f32_to_f16<<<512,  256, 0, stream>>>(Wv, Wvh, (int)DD);
  cvt_f32_to_f16<<<512,  256, 0, stream>>>(Wo, Woh, (int)DD);

  // projections: (M/16) * (N/64) = 256 * 16 = 4096 waves = 512 blocks x 8 waves
  const float kScaling = 0.125f;                 // Dh^-0.5 = 64^-0.5
  proj_gemm<<<512, 256, 0, stream>>>(Xq,  Wqh, bq, Qh, kScaling, 0);
  proj_gemm<<<512, 256, 0, stream>>>(Xkv, Wkh, bk, Kh, 1.0f,     0);
  proj_gemm<<<512, 256, 0, stream>>>(Xkv, Wvh, bv, Vt, 1.0f,     1);

  // attention: B*H*(S/16) = 2*16*128 = 4096 waves, one per block
  flash_attn<<<4096, 32, 0, stream>>>(Qh, Kh, Vt, msk, AO);

  // output projection -> f32 d_out
  proj_gemm<<<512, 256, 0, stream>>>(AO, Woh, bo, (float*)d_out, 1.0f, 2);
}
